// MultiQueryAttentionBatched_25486335935234
// MI455X (gfx1250) — matrix operand order repaired and checked by reading, not compiled, not run on hardware
//
#include <hip/hip_runtime.h>

// ---------------------------------------------------------------------------
// MQA forward for MI455X (gfx1250, wave32, WMMA).
// D_MODEL=1024, H=16, KD=VD=64, B=4, N=M_LEN=2048.
// All matmuls via v_wmma_f32_16x16x32_bf16 (f32 accum).
// ---------------------------------------------------------------------------

typedef __bf16 v16bf __attribute__((ext_vector_type(16)));
typedef float  v8f   __attribute__((ext_vector_type(8)));

#define Bq   4
#define Hh   16
#define Nn   2048
#define Mm   2048
#define Dd   1024
#define KDv  64
#define VDv  64

// ---- WMMA fragment loaders (layouts per CDNA5 ISA §7.12.2) ------
// A (16x32 bf16): lane r=l&15 holds row r; lanes<16: K {0..7,16..23}+k0,
//                 lanes>=16: K {8..15,24..31}+k0. Element e -> vgpr e/2.
__device__ __forceinline__ v16bf load_a_f32(const float* base, int ld, int k0, int lane) {
    int r = lane & 15, hh = lane >> 4;
    const float* p = base + (size_t)r * ld + k0 + hh * 8;
    v16bf out;
#pragma unroll
    for (int i = 0; i < 8; ++i) out[i] = (__bf16)p[i];
#pragma unroll
    for (int i = 0; i < 8; ++i) out[8 + i] = (__bf16)p[16 + i];
    return out;
}
__device__ __forceinline__ v16bf load_a_bf16(const __bf16* base, int ld, int k0, int lane) {
    int r = lane & 15, hh = lane >> 4;
    const __bf16* p = base + (size_t)r * ld + k0 + hh * 8;
    v16bf out;
#pragma unroll
    for (int i = 0; i < 8; ++i) out[i] = p[i];
#pragma unroll
    for (int i = 0; i < 8; ++i) out[8 + i] = p[16 + i];
    return out;
}
// B (32x16 bf16): lane holds column n = (l&15); lanes<16: K {0..7,16..23}+k0,
// lanes>=16: K {8..15,24..31}+k0, stored as W[n][K] (row-major over n).
__device__ __forceinline__ v16bf load_b_bf16(const __bf16* base, int ld, int k0, int lane) {
    int r = lane & 15, hh = lane >> 4;
    const __bf16* p = base + (size_t)r * ld + k0 + hh * 8;
    v16bf out;
#pragma unroll
    for (int i = 0; i < 8; ++i) out[i] = p[i];
#pragma unroll
    for (int i = 0; i < 8; ++i) out[8 + i] = p[16 + i];
    return out;
}

__device__ __forceinline__ v8f wmma_bf16(v16bf a, v16bf b, v8f c) {
    return __builtin_amdgcn_wmma_f32_16x16x32_bf16(false, a, false, b, (short)0, c, false, false);
}

// ---- weight repack kernels (f32 -> bf16, B-operand friendly layouts) -------
__global__ void pack_wq(const float* __restrict__ Pq, __bf16* __restrict__ W) {
    // W[(h*64+k)][d] = Pq[h][d][k]   (1024 x 1024)
    int i = blockIdx.x * blockDim.x + threadIdx.x;
    int d = i & 1023, hk = i >> 10, h = hk >> 6, k = hk & 63;
    W[i] = (__bf16)Pq[((size_t)h * Dd + d) * KDv + k];
}
__global__ void pack_wkv(const float* __restrict__ Pk, const float* __restrict__ Pv,
                         __bf16* __restrict__ W) {
    // W[n][d], n<64 -> Pk[d][n], n>=64 -> Pv[d][n-64]   (128 x 1024)
    int i = blockIdx.x * blockDim.x + threadIdx.x;
    int d = i & 1023, n = i >> 10;
    float v = (n < 64) ? Pk[(size_t)d * KDv + n] : Pv[(size_t)d * VDv + (n - 64)];
    W[i] = (__bf16)v;
}
__global__ void pack_wo(const float* __restrict__ Po, __bf16* __restrict__ W) {
    // W[d][h*64+v] = Po[h][d][v]   (1024 x 1024)
    int i = blockIdx.x * blockDim.x + threadIdx.x;
    int hv = i & 1023, d = i >> 10, h = hv >> 6, v = hv & 63;
    W[i] = (__bf16)Po[((size_t)h * Dd + d) * VDv + v];
}

// ---- WMMA GEMM, 64x16 output tile per wave (4 row-tiles share one B frag) --
#define MODE_Q  0   // out bf16: Q[b][h][n][kd]
#define MODE_KV 1   // out bf16: K[b][m][kd] ; V^T[b][vd][m]
#define MODE_Y  2   // out f32 : row-major
__global__ __launch_bounds__(32)
void gemm_bf16(const float* __restrict__ A, int lda,
               const __bf16* __restrict__ W, int ldw, int Kdim,
               void* __restrict__ out, int mode) {
    int lane = threadIdx.x;
    int row0 = blockIdx.x * 64;
    int n0   = blockIdx.y * 16;
    v8f acc[4] = {};
    for (int k0 = 0; k0 < Kdim; k0 += 32) {
        v16bf b = load_b_bf16(W + (size_t)n0 * ldw, ldw, k0, lane);   // loaded once
#pragma unroll
        for (int t = 0; t < 4; ++t) {
            v16bf a = load_a_f32(A + (size_t)(row0 + 16 * t) * lda, lda, k0, lane);
            acc[t] = wmma_bf16(a, b, acc[t]);
        }
    }
    int r = lane & 15, hh = lane >> 4;
#pragma unroll
    for (int t = 0; t < 4; ++t) {
#pragma unroll
        for (int i = 0; i < 8; ++i) {
            int row = row0 + 16 * t + i + 8 * hh;  // C layout: vgpr i -> M=i (+8 hi half)
            int c   = n0 + r;
            float val = acc[t][i];
            if (mode == MODE_Q) {
                __bf16* Q = (__bf16*)out;
                int b = row >> 11, n = row & 2047, head = c >> 6, k = c & 63;
                Q[(((size_t)b * Hh + head) * Nn + n) * KDv + k] = (__bf16)val;
            } else if (mode == MODE_KV) {
                __bf16* Kb = (__bf16*)out;
                __bf16* Vt = Kb + (size_t)Bq * Mm * KDv;
                int b = row >> 11, m = row & 2047;
                if (c < 64) Kb[((size_t)b * Mm + m) * KDv + c] = (__bf16)val;
                else        Vt[((size_t)b * VDv + (c - 64)) * Mm + m] = (__bf16)val;
            } else {
                ((float*)out)[(size_t)row * Dd + c] = val;
            }
        }
    }
}

// ---- flash attention: one wave = 16 queries x (all 2048 keys) x 64 vdim ----
__global__ __launch_bounds__(32)
void mqa_attn(const __bf16* __restrict__ Qb, const __bf16* __restrict__ Kb,
              const __bf16* __restrict__ Vt, const float* __restrict__ mask,
              float* __restrict__ Of) {
    __shared__ __bf16 sP[16 * 32];      // 16 queries x 32 keys of probabilities
    int lane = threadIdx.x;
    int idx = blockIdx.x;               // b*H*128 + h*128 + nt
    int nt = idx & 127, bh = idx >> 7, h = bh & 15, b = bh >> 4;
    int n0 = nt << 4;
    int r = lane & 15, hh = lane >> 4;

    const __bf16* Qtile = Qb + (((size_t)b * Hh + h) * Nn + n0) * KDv;
    v16bf qa0 = load_a_bf16(Qtile, KDv, 0, lane);    // kd 0..31
    v16bf qa1 = load_a_bf16(Qtile, KDv, 32, lane);   // kd 32..63

    const __bf16* Kbb = Kb + (size_t)b * Mm * KDv;
    const __bf16* Vbb = Vt + (size_t)b * VDv * Mm;
    const float*  mk  = mask + ((size_t)b * Nn + n0) * Mm;

    v8f o0 = {}, o1 = {}, o2 = {}, o3 = {};
    float mrun[8], lsum[8];
#pragma unroll
    for (int i = 0; i < 8; ++i) { mrun[i] = -3.0e38f; lsum[i] = 0.0f; }

    for (int mb = 0; mb < Mm; mb += 32) {
        // prefetch next 32-key K block (32*64 bf16 = 4 KiB; lane stride 128 B)
        // and next V^T block (2 rows/lane) -> global_prefetch_b8, overlaps softmax
        if (mb + 32 < Mm) {
            __builtin_prefetch(Kbb + (size_t)(mb + 32) * KDv + (size_t)lane * 64, 0, 3);
            __builtin_prefetch(Vbb + (size_t)(lane * 2 + 0) * Mm + mb + 32, 0, 3);
            __builtin_prefetch(Vbb + (size_t)(lane * 2 + 1) * Mm + mb + 32, 0, 3);
        }
        // S = Q @ K^T : two 16x16 key tiles, K-dim (=kd 64) in 2 wmma steps
        v16bf k00 = load_b_bf16(Kbb + (size_t)mb * KDv,        KDv, 0,  lane);
        v16bf k01 = load_b_bf16(Kbb + (size_t)mb * KDv,        KDv, 32, lane);
        v16bf k10 = load_b_bf16(Kbb + (size_t)(mb + 16) * KDv, KDv, 0,  lane);
        v16bf k11 = load_b_bf16(Kbb + (size_t)(mb + 16) * KDv, KDv, 32, lane);
        v8f z = {};
        v8f s0 = wmma_bf16(qa0, k00, z); s0 = wmma_bf16(qa1, k01, s0);
        v8f s1 = wmma_bf16(qa0, k10, z); s1 = wmma_bf16(qa1, k11, s1);

        // online softmax per row (row = i + 8*hh; stats shared by 16-lane half)
#pragma unroll
        for (int i = 0; i < 8; ++i) {
            int row = i + 8 * hh;
            const float* mp = mk + (size_t)row * Mm + mb;
            float v0 = s0[i] + mp[r];
            float v1 = s1[i] + mp[16 + r];
            float t = fmaxf(v0, v1);
#pragma unroll
            for (int off = 1; off < 16; off <<= 1) t = fmaxf(t, __shfl_xor(t, off, 32));
            float mnew  = fmaxf(mrun[i], t);
            float alpha = __expf(mrun[i] - mnew);
            mrun[i] = mnew;
            float p0 = __expf(v0 - mnew), p1 = __expf(v1 - mnew);
            float rs = p0 + p1;
#pragma unroll
            for (int off = 1; off < 16; off <<= 1) rs += __shfl_xor(rs, off, 32);
            lsum[i] = lsum[i] * alpha + rs;
            o0[i] *= alpha; o1[i] *= alpha; o2[i] *= alpha; o3[i] *= alpha;
            sP[row * 32 + r]      = (__bf16)p0;     // C layout -> LDS (row-major 16x32)
            sP[row * 32 + 16 + r] = (__bf16)p1;
        }
        __syncthreads();
        // re-read probabilities in A-operand layout (row = r, K = keys)
        v16bf pa;
        {
            int kb = hh * 8;
#pragma unroll
            for (int i = 0; i < 8; ++i) pa[i] = sP[r * 32 + kb + i];
#pragma unroll
            for (int i = 0; i < 8; ++i) pa[8 + i] = sP[r * 32 + 16 + kb + i];
        }
        __syncthreads();
        // O += P @ V : B operand from V^T[b][vd][m] (contiguous in m)
        v16bf vb;
        vb = load_b_bf16(Vbb + (size_t)0  * Mm + mb, Mm, 0, lane); o0 = wmma_bf16(pa, vb, o0);
        vb = load_b_bf16(Vbb + (size_t)16 * Mm + mb, Mm, 0, lane); o1 = wmma_bf16(pa, vb, o1);
        vb = load_b_bf16(Vbb + (size_t)32 * Mm + mb, Mm, 0, lane); o2 = wmma_bf16(pa, vb, o2);
        vb = load_b_bf16(Vbb + (size_t)48 * Mm + mb, Mm, 0, lane); o3 = wmma_bf16(pa, vb, o3);
    }

    // normalize and store O in [b][n][h*64+v] layout (f32) for the Y GEMM
#pragma unroll
    for (int i = 0; i < 8; ++i) {
        int row = i + 8 * hh;
        float inv = 1.0f / lsum[i];
        size_t base = ((size_t)b * Nn + n0 + row) * (Hh * VDv) + h * VDv;
        Of[base +  0 + r] = o0[i] * inv;
        Of[base + 16 + r] = o1[i] * inv;
        Of[base + 32 + r] = o2[i] * inv;
        Of[base + 48 + r] = o3[i] * inv;
    }
}

// ---------------------------------------------------------------------------
extern "C" void kernel_launch(void* const* d_in, const int* in_sizes, int n_in,
                              void* d_out, int out_size, void* d_ws, size_t ws_size,
                              hipStream_t stream) {
    (void)in_sizes; (void)n_in; (void)out_size; (void)ws_size;
    const float* X    = (const float*)d_in[0];
    const float* Mi   = (const float*)d_in[1];
    const float* mask = (const float*)d_in[2];
    const float* Pq   = (const float*)d_in[3];
    const float* Pk   = (const float*)d_in[4];
    const float* Pv   = (const float*)d_in[5];
    const float* Po   = (const float*)d_in[6];
    float* Y = (float*)d_out;

    // workspace layout (bytes)
    char* ws = (char*)d_ws;
    __bf16* Wq  = (__bf16*)(ws);                                //  2 MiB: 1024x1024
    __bf16* Wkv = (__bf16*)(ws + 2097152);                      //  256 KiB: 128x1024
    __bf16* Wo  = (__bf16*)(ws + 2097152 + 262144);             //  2 MiB: 1024x1024
    __bf16* Qb  = (__bf16*)(ws + 4456448);                      // 16 MiB: B*H*N*64
    __bf16* KV  = (__bf16*)(ws + 4456448 + 16777216);           //  2 MiB: K + V^T
    float*  Of  = (float*) (ws + 4456448 + 16777216 + 2097152); // 32 MiB: B*N*1024
    __bf16* Kb = KV;
    __bf16* Vt = KV + (size_t)Bq * Mm * KDv;

    // 1) repack weights to bf16
    pack_wq <<<4096, 256, 0, stream>>>(Pq, Wq);
    pack_wkv<<< 512, 256, 0, stream>>>(Pk, Pv, Wkv);
    pack_wo <<<4096, 256, 0, stream>>>(Po, Wo);

    // 2) projections: Q = X @ Wq ; [K|V] = M @ Wkv   (64x16 tile per wave)
    gemm_bf16<<<dim3(128, 64), 32, 0, stream>>>(X,  Dd, Wq,  Dd, Dd, (void*)Qb, MODE_Q);
    gemm_bf16<<<dim3(128,  8), 32, 0, stream>>>(Mi, Dd, Wkv, Dd, Dd, (void*)KV, MODE_KV);

    // 3) flash attention core
    mqa_attn<<<Bq * Hh * (Nn / 16), 32, 0, stream>>>(Qb, Kb, Vt, mask, Of);

    // 4) output projection: Y = O_flat @ Wo
    gemm_bf16<<<dim3(128, 64), 32, 0, stream>>>(Of, Hh * VDv, Wo, Dd, Dd, (void*)Y, MODE_Y);
}
